// ET_convGRU_28595892257052
// MI455X (gfx1250) — matrix operand order repaired and checked by reading, not compiled, not run on hardware
//
#include <hip/hip_runtime.h>
#include <hip/hip_bf16.h>
#include <math.h>

#define BB   128
#define TT   16
#define CC   512
#define HHH  6
#define WWW  6
#define HID  512
#define NPIX (BB * HHH * WWW)   /* 4608 */
#define KDIM (CC + HID)         /* 1024 */

typedef __attribute__((ext_vector_type(16))) _Float16 v16h;
typedef __attribute__((ext_vector_type(8)))  _Float16 h8;
typedef __attribute__((ext_vector_type(8)))  float    v8f;

// fast activations (v_exp_f32 + v_rcp_f32, no IEEE divide sequence)
__device__ __forceinline__ float fast_sigmoid(float t) {
    return __builtin_amdgcn_rcpf(1.0f + __expf(-t));
}
__device__ __forceinline__ float fast_tanh(float t) {
    // tanh(t) = 1 - 2/(1+exp(2t)) ; exp->inf => rcp->0 => +1 ; exp->0 => -1
    return 1.0f - 2.0f * __builtin_amdgcn_rcpf(1.0f + __expf(2.0f * t));
}

// ---------------------------------------------------------------------------
// small elementwise kernels
// ---------------------------------------------------------------------------
__global__ void k_f32_to_f16(const float* __restrict__ src,
                             _Float16* __restrict__ dst, int n) {
    int i = blockIdx.x * 256 + threadIdx.x;
    if (i < n) dst[i] = (_Float16)src[i];
}

__global__ void k_zero2(float* __restrict__ a, float* __restrict__ b, int n) {
    int i = blockIdx.x * 256 + threadIdx.x;
    if (i < n) { a[i] = 0.0f; b[i] = 0.0f; }
}

// layer-0 input prep for step t: x16[p][c] = (f16)x[b][t][c][hw], h16 = (f16)h0
__global__ void k_fill0(const float* __restrict__ x, int t,
                        const float* __restrict__ h0,
                        _Float16* __restrict__ x16,
                        _Float16* __restrict__ h16) {
    int i = blockIdx.x * 256 + threadIdx.x;     // i < NPIX*CC
    if (i >= NPIX * CC) return;
    int p   = i >> 9;          // /512
    int c   = i & 511;
    int b   = p / 36;
    int r36 = p - b * 36;
    x16[i] = (_Float16)x[(((size_t)b * TT + t) * CC + c) * 36 + r36];
    h16[i] = (_Float16)h0[i];
}

__global__ void k_fill1(const float* __restrict__ h1,
                        _Float16* __restrict__ h16) {
    int i = blockIdx.x * 256 + threadIdx.x;
    if (i < NPIX * HID) h16[i] = (_Float16)h1[i];
}

// out[b][c][hw] = h1[p][c] + x0[b][c][hw]
__global__ void k_final(const float* __restrict__ h1,
                        const float* __restrict__ x0,
                        float* __restrict__ out) {
    int i = blockIdx.x * 256 + threadIdx.x;     // i < BB*HID*36
    if (i >= BB * HID * 36) return;
    int b   = i / (HID * 36);
    int rem = i - b * HID * 36;
    int c   = rem / 36;
    int r36 = rem - c * 36;
    int p   = b * 36 + r36;
    out[i] = h1[(size_t)p * HID + c] + x0[i];
}

// ---------------------------------------------------------------------------
// Shared GEMM mainloop.
//  - Block 256 threads (8 waves), block tile 128 rows x 128 cols, k-step 32.
//  - A/B tiles staged in LDS in WMMA *fragment order* [tile][lane][16 halves]:
//      A (16x32): lane L row L&15; halves i -> K=(L>=16?16:0)+i.
//      B (32x16): lane L col L&15; halves i -> K=(L>=16?16:0)+i.
//    so every fragment fetch is a single aligned 32B ds read.
//  - Next k-slice register-prefetched from global between barriers.
// ---------------------------------------------------------------------------
__device__ __forceinline__ void gemm_mainloop(const _Float16* __restrict__ Alo,
                                              const _Float16* __restrict__ Ahi,
                                              const _Float16* __restrict__ W,
                                              _Float16* ldsA, _Float16* ldsB,
                                              int rowBlk, int colBlk,
                                              v8f acc[4][2]) {
    const int tid   = threadIdx.x;
    const int lane  = tid & 31;
    const int wave  = tid >> 5;
    const int waveRow = wave >> 2;     // 0..1  (64-row wave tile)
    const int waveCol = wave & 3;      // 0..3  (32-col wave tile)

    // staging assignment: thread -> (row/col srow 0..127, k-half pair 0/1)
    const int srow  = tid >> 1;
    const int spair = tid & 1;
    const int atile = srow >> 4;
    const int alane = srow & 15;
    _Float16* stA0 = ldsA + atile * 512 + (alane + spair * 16) * 16;     // k 0..7 of chunk -> halves 0..7
    _Float16* stA1 = ldsA + atile * 512 + (alane + spair * 16) * 16 + 8; // k 8..15 of chunk -> halves 8..15
    _Float16* stB  = ldsB + atile * 512 + (alane + spair * 16) * 16;

    const _Float16* gArow = /*A base set per-k*/ nullptr;
    const _Float16* gBrow = W + (size_t)(colBlk + srow) * KDIM + spair * 16;

    // prefetch k-slice 0
    h8  raLo, raHi;
    v16h rb;
    {
        const _Float16* gA = Alo + (size_t)(rowBlk + srow) * CC + spair * 16;
        raLo = *(const h8*)(gA);
        raHi = *(const h8*)(gA + 8);
        rb   = *(const v16h*)(gBrow);
    }
    (void)gArow;

    for (int kk = 0; kk < KDIM; kk += 32) {
        __syncthreads();                       // previous compute done reading LDS
        *(h8*)stA0  = raLo;
        *(h8*)stA1  = raHi;
        *(v16h*)stB = rb;
        __syncthreads();

        const int kn = kk + 32;
        if (kn < KDIM) {                       // prefetch next slice
            const _Float16* A  = (kn < CC) ? Alo : Ahi;
            const int k0 = kn & (CC - 1);
            const _Float16* gA = A + (size_t)(rowBlk + srow) * CC + k0 + spair * 16;
            raLo = *(const h8*)(gA);
            raHi = *(const h8*)(gA + 8);
            rb   = *(const v16h*)(gBrow + kn);
        }

        v16h a[4], b[2];
#pragma unroll
        for (int i = 0; i < 4; ++i)
            a[i] = *(const v16h*)(ldsA + (waveRow * 4 + i) * 512 + lane * 16);
#pragma unroll
        for (int j = 0; j < 2; ++j)
            b[j] = *(const v16h*)(ldsB + (waveCol * 2 + j) * 512 + lane * 16);
#pragma unroll
        for (int i = 0; i < 4; ++i)
#pragma unroll
            for (int j = 0; j < 2; ++j)
                acc[i][j] = __builtin_amdgcn_wmma_f32_16x16x32_f16(
                    false, a[i], false, b[j], (short)0, acc[i][j], false, false);
    }
}

// gates GEMM: D(4608x1024) = [Alo|Ahi] x Wg^T ; sigmoid epilogue.
// cols 0..511 -> rh = sigmoid(.)*h (f16) ; cols 512..1023 -> z = sigmoid(.)
__global__ __launch_bounds__(256)
void k_gemm_gates(const _Float16* __restrict__ Alo,
                  const _Float16* __restrict__ Ahi,
                  const _Float16* __restrict__ W,
                  const float* __restrict__ bias,
                  const float* __restrict__ h,
                  _Float16* __restrict__ rh,
                  float* __restrict__ z) {
    __shared__ _Float16 ldsA[8 * 512];
    __shared__ _Float16 ldsB[8 * 512];
    const int lane = threadIdx.x & 31;
    const int wave = threadIdx.x >> 5;
    const int lr   = lane & 15;
    const int lhi  = lane >> 4;
    const int rowBase = blockIdx.x * 128 + (wave >> 2) * 64;
    const int colBase = blockIdx.y * 128 + (wave & 3) * 32;

    v8f acc[4][2];
#pragma unroll
    for (int i = 0; i < 4; ++i)
#pragma unroll
        for (int j = 0; j < 2; ++j) acc[i][j] = (v8f)(0.0f);

    gemm_mainloop(Alo, Ahi, W, ldsA, ldsB, blockIdx.x * 128, blockIdx.y * 128, acc);

#pragma unroll
    for (int j = 0; j < 2; ++j) {
        const int col = colBase + j * 16 + lr;
        const float bv = bias[col];
#pragma unroll
        for (int i = 0; i < 4; ++i) {
#pragma unroll
            for (int v = 0; v < 8; ++v) {
                const int row = rowBase + i * 16 + v + (lhi << 3);
                const float g = fast_sigmoid(acc[i][j][v] + bv);
                if (col < HID) {
                    const size_t idx = (size_t)row * HID + col;
                    rh[idx] = (_Float16)(g * h[idx]);
                } else {
                    z[(size_t)row * HID + (col - HID)] = g;
                }
            }
        }
    }
}

// candidate GEMM: D(4608x512) = [Alo|Arh] x Wc^T ; tanh + GRU update epilogue.
__global__ __launch_bounds__(256)
void k_gemm_cand(const _Float16* __restrict__ Alo,
                 const _Float16* __restrict__ Arh,
                 const _Float16* __restrict__ W,
                 const float* __restrict__ bias,
                 const float* __restrict__ z,
                 float* __restrict__ h,
                 _Float16* __restrict__ y16) {
    __shared__ _Float16 ldsA[8 * 512];
    __shared__ _Float16 ldsB[8 * 512];
    const int lane = threadIdx.x & 31;
    const int wave = threadIdx.x >> 5;
    const int lr   = lane & 15;
    const int lhi  = lane >> 4;
    const int rowBase = blockIdx.x * 128 + (wave >> 2) * 64;
    const int colBase = blockIdx.y * 128 + (wave & 3) * 32;

    v8f acc[4][2];
#pragma unroll
    for (int i = 0; i < 4; ++i)
#pragma unroll
        for (int j = 0; j < 2; ++j) acc[i][j] = (v8f)(0.0f);

    gemm_mainloop(Alo, Arh, W, ldsA, ldsB, blockIdx.x * 128, blockIdx.y * 128, acc);

#pragma unroll
    for (int j = 0; j < 2; ++j) {
        const int col = colBase + j * 16 + lr;
        const float bv = bias[col];
#pragma unroll
        for (int i = 0; i < 4; ++i) {
#pragma unroll
            for (int v = 0; v < 8; ++v) {
                const int row = rowBase + i * 16 + v + (lhi << 3);
                const size_t idx = (size_t)row * HID + col;
                const float nv = fast_tanh(acc[i][j][v] + bv);
                const float zz = z[idx];
                const float hn = (1.0f - zz) * h[idx] + zz * nv;
                h[idx] = hn;
                if (y16) y16[idx] = (_Float16)hn;
            }
        }
    }
}

// ---------------------------------------------------------------------------
extern "C" void kernel_launch(void* const* d_in, const int* in_sizes, int n_in,
                              void* d_out, int out_size, void* d_ws, size_t ws_size,
                              hipStream_t stream) {
    const float* x   = (const float*)d_in[0];
    const float* x0  = (const float*)d_in[1];
    const float* Wg0 = (const float*)d_in[2];
    const float* bg0 = (const float*)d_in[3];
    const float* Wc0 = (const float*)d_in[4];
    const float* bc0 = (const float*)d_in[5];
    const float* Wg1 = (const float*)d_in[6];
    const float* bg1 = (const float*)d_in[7];
    const float* Wc1 = (const float*)d_in[8];
    const float* bc1 = (const float*)d_in[9];
    float* out = (float*)d_out;

    char* ws = (char*)d_ws;
    size_t off = 0;
    auto carve = [&](size_t bytes) -> char* {
        char* p = ws + off;
        off += (bytes + 255) & ~(size_t)255;
        return p;
    };

    _Float16* Wg0h = (_Float16*)carve((size_t)2 * HID * KDIM * 2);
    _Float16* Wc0h = (_Float16*)carve((size_t)HID * KDIM * 2);
    _Float16* Wg1h = (_Float16*)carve((size_t)2 * HID * KDIM * 2);
    _Float16* Wc1h = (_Float16*)carve((size_t)HID * KDIM * 2);
    _Float16* x16  = (_Float16*)carve((size_t)NPIX * CC * 2);   // layer-0 x_t
    _Float16* h16  = (_Float16*)carve((size_t)NPIX * HID * 2);  // current layer's h (f16)
    _Float16* y16  = (_Float16*)carve((size_t)NPIX * HID * 2);  // layer-0 output (layer-1 input)
    _Float16* rh16 = (_Float16*)carve((size_t)NPIX * HID * 2);  // r*h (f16)
    float*    zbuf = (float*)carve((size_t)NPIX * HID * 4);
    float*    h0   = (float*)carve((size_t)NPIX * HID * 4);
    float*    h1   = (float*)carve((size_t)NPIX * HID * 4);

    const int EW  = NPIX * HID;                 // 2,359,296 elementwise size
    const int EWB = (EW + 255) / 256;

    // weights -> f16 (deterministic, once per launch)
    k_f32_to_f16<<<(2 * HID * KDIM + 255) / 256, 256, 0, stream>>>(Wg0, Wg0h, 2 * HID * KDIM);
    k_f32_to_f16<<<(HID * KDIM + 255) / 256, 256, 0, stream>>>(Wc0, Wc0h, HID * KDIM);
    k_f32_to_f16<<<(2 * HID * KDIM + 255) / 256, 256, 0, stream>>>(Wg1, Wg1h, 2 * HID * KDIM);
    k_f32_to_f16<<<(HID * KDIM + 255) / 256, 256, 0, stream>>>(Wc1, Wc1h, HID * KDIM);
    k_zero2<<<EWB, 256, 0, stream>>>(h0, h1, EW);

    const dim3 gGates(NPIX / 128, (2 * HID) / 128);  // 36 x 8
    const dim3 gCand (NPIX / 128, HID / 128);        // 36 x 4

    for (int t = 0; t < TT; ++t) {
        // ---- layer 0 ----
        k_fill0<<<EWB, 256, 0, stream>>>(x, t, h0, x16, h16);
        k_gemm_gates<<<gGates, 256, 0, stream>>>(x16, h16, Wg0h, bg0, h0, rh16, zbuf);
        k_gemm_cand <<<gCand, 256, 0, stream>>>(x16, rh16, Wc0h, bc0, zbuf, h0, y16);
        // ---- layer 1 (consumes y16 = layer-0 output at step t) ----
        k_fill1<<<EWB, 256, 0, stream>>>(h1, h16);
        k_gemm_gates<<<gGates, 256, 0, stream>>>(y16, h16, Wg1h, bg1, h1, rh16, zbuf);
        k_gemm_cand <<<gCand, 256, 0, stream>>>(y16, rh16, Wc1h, bc1, zbuf, h1, (_Float16*)nullptr);
    }

    k_final<<<EWB, 256, 0, stream>>>(h1, x0, out);
}
